// EdgeAttrGATModel_644245095201
// MI455X (gfx1250) — matrix operand order repaired and checked by reading, not compiled, not run on hardware
//
#include <hip/hip_runtime.h>
#include <hip/hip_bf16.h>

typedef __attribute__((ext_vector_type(16))) _Float16 v16h;
typedef __attribute__((ext_vector_type(8)))  float    v8f;

#define HIDC 128
#define NSLOPE 0.2f

// ---------------- utility fills ----------------
__global__ void k_fill_f32(float* p, float v, int n) {
    int i = blockIdx.x * blockDim.x + threadIdx.x;
    if (i < n) p[i] = v;
}
__global__ void k_fill_u32(unsigned* p, unsigned v, int n) {
    int i = blockIdx.x * blockDim.x + threadIdx.x;
    if (i < n) p[i] = v;
}

// ---------------- monotone float<->uint encoding for atomicMax ----------------
__device__ __forceinline__ unsigned enc_f(float f) {
    unsigned u = __float_as_uint(f);
    return (u & 0x80000000u) ? ~u : (u | 0x80000000u);
}
__device__ __forceinline__ float dec_f(unsigned u) {
    return __uint_as_float((u & 0x80000000u) ? (u & 0x7FFFFFFFu) : ~u);
}

// ---------------- convert weights f32 -> f16 ----------------
__global__ void k_f32_to_f16(const float* __restrict__ in, _Float16* __restrict__ out, int n) {
    int i = blockIdx.x * blockDim.x + threadIdx.x;
    if (i < n) out[i] = (_Float16)in[i];
}

// ---------------- embedding gather -> f16 ----------------
__global__ void k_embed(const int* __restrict__ xidx, const float* __restrict__ tab,
                        _Float16* __restrict__ out, int n) {
    int i = blockIdx.x * blockDim.x + threadIdx.x;   // over n*128
    if (i >= n * HIDC) return;
    int node = i >> 7, c = i & 127;
    out[i] = (_Float16)tab[(size_t)xidx[node] * HIDC + c];
}

// ---------------- relu(f32) -> f16 (feed next layer's GEMM) ----------------
__global__ void k_relu_tohalf(const float* __restrict__ in, _Float16* __restrict__ out, int n) {
    int i = blockIdx.x * blockDim.x + threadIdx.x;
    if (i < n) out[i] = (_Float16)fmaxf(in[i], 0.0f);
}

// ---------------- H = X @ W^T via WMMA (wave32, 16x16 tile per wave) ----------------
// X: [nrows,128] f16 row-major.  Wh: [128,128] f16 row-major (H[n,o]=sum_c X[n,c]*W[o,c])
__global__ void k_gemm_wmma(const _Float16* __restrict__ X, const _Float16* __restrict__ Wh,
                            float* __restrict__ Hout, int nrows) {
    int wave = threadIdx.x >> 5;
    int lane = threadIdx.x & 31;
    int tile = blockIdx.x * 8 + wave;           // 8 waves per 256-thread block
    int ntr  = nrows >> 4;
    int tr   = tile >> 3;                       // row tile
    int tc   = tile & 7;                        // col tile (128/16 = 8)
    if (tr >= ntr) return;                      // wave-uniform: EXEC stays all-ones
    int m0 = tr << 4, n0 = tc << 4;
    bool hi = lane >= 16;
    int l15 = lane & 15;
    int arow = m0 + l15;
    int bcol = n0 + l15;

    v8f acc = {};
    #pragma unroll
    for (int k0 = 0; k0 < 128; k0 += 32) {
        union { v16h v; _Float16 h[16]; } a, b;
        // A fragment: lanes 0-15 -> K {k0..k0+15}; lanes 16-31 -> +16
        const _Float16* ap = X + (size_t)arow * HIDC + k0 + (hi ? 16 : 0);
        #pragma unroll
        for (int i = 0; i < 8; ++i) { a.h[i] = ap[i]; a.h[i + 8] = ap[8 + i]; }
        // B fragment: lane = column, contiguous K block (lanes 16-31 -> K+16)
        const _Float16* bp = Wh + (size_t)bcol * HIDC + k0 + (hi ? 16 : 0);
        #pragma unroll
        for (int i = 0; i < 16; ++i) b.h[i] = bp[i];
        acc = __builtin_amdgcn_wmma_f32_16x16x32_f16(false, a.v, false, b.v,
                                                     (short)0, acc, false, false);
    }
    // C/D layout: lane<16 -> VGPR r = (M=r, N=lane); lane>=16 -> (M=r+8, N=lane-16)
    int rbase = m0 + (hi ? 8 : 0);
    #pragma unroll
    for (int r = 0; r < 8; ++r)
        Hout[(size_t)(rbase + r) * HIDC + n0 + l15] = acc[r];
}

// ---------------- per-node attention scalars: s=H.a_src, d=H.a_dst ----------------
__global__ void k_node_scores(const float* __restrict__ H, const float* __restrict__ asrc,
                              const float* __restrict__ adst,
                              float* __restrict__ so, float* __restrict__ dof, int n) {
    int wave = threadIdx.x >> 5, lane = threadIdx.x & 31;
    int node = blockIdx.x * 8 + wave;
    if (node >= n) return;
    const float* h = H + (size_t)node * HIDC;
    float ss = 0.f, dd = 0.f;
    #pragma unroll
    for (int c = lane; c < HIDC; c += 32) { float v = h[c]; ss += v * asrc[c]; dd += v * adst[c]; }
    #pragma unroll
    for (int off = 16; off; off >>= 1) {
        ss += __shfl_down(ss, off, 32);
        dd += __shfl_down(dd, off, 32);
    }
    if (lane == 0) { so[node] = ss; dof[node] = dd; }
}

// ---------------- edge logits + segment max ----------------
__global__ void k_edge_logits(const int* __restrict__ src, const int* __restrict__ dst,
                              const float* __restrict__ ea,
                              const float* __restrict__ as_, const float* __restrict__ ad_,
                              float* __restrict__ alphaE, unsigned* __restrict__ segmax,
                              int E_, int n) {
    int e = blockIdx.x * blockDim.x + threadIdx.x;
    int total = E_ + n;
    if (e >= total) return;
    int s, d; float w;
    if (e < E_) { s = src[e]; d = dst[e]; w = ea[e]; }
    else        { s = d = e - E_; w = 1.0f; }
    float l = ad_[d] + as_[s];
    l = (l >= 0.f) ? l : NSLOPE * l;       // LeakyReLU(0.2)
    l *= w;                                // multiplicative edge attr
    alphaE[e] = l;
    atomicMax(&segmax[d], enc_f(l));
}

// ---------------- exp + segment sum ----------------
__global__ void k_edge_expsum(const int* __restrict__ dst, float* __restrict__ alphaE,
                              const unsigned* __restrict__ segmax, float* __restrict__ segsum,
                              int E_, int n) {
    int e = blockIdx.x * blockDim.x + threadIdx.x;
    int total = E_ + n;
    if (e >= total) return;
    int d = (e < E_) ? dst[e] : (e - E_);
    float ex = __expf(alphaE[e] - dec_f(segmax[d]));
    alphaE[e] = ex;
    atomicAdd(&segsum[d], ex);
}

// ---------------- weighted aggregation: OUT[dst] += H[src]*alpha (wave per edge) ----------------
__global__ void k_edge_aggregate(const int* __restrict__ src, const int* __restrict__ dst,
                                 const float* __restrict__ alphaE, const float* __restrict__ segsum,
                                 const float* __restrict__ H, float* __restrict__ OUT,
                                 int E_, int n) {
    int gid  = blockIdx.x * blockDim.x + threadIdx.x;
    int wave = gid >> 5;
    int lane = threadIdx.x & 31;
    int total = E_ + n;
    if (wave >= total) return;
    int s, d;
    if (wave < E_) { s = src[wave]; d = dst[wave]; } else { s = d = wave - E_; }
    float w = alphaE[wave] / (segsum[d] + 1e-16f);
    const float4* hp = (const float4*)(H + (size_t)s * HIDC);
    float4 v = hp[lane];
    float* o = OUT + (size_t)d * HIDC + lane * 4;
    atomicAdd(o + 0, v.x * w);
    atomicAdd(o + 1, v.y * w);
    atomicAdd(o + 2, v.z * w);
    atomicAdd(o + 3, v.w * w);
}

// ---------------- relu + mean-pool accumulation (wave per node) ----------------
__global__ void k_pool(const float* __restrict__ OUT2, const int* __restrict__ batch,
                       float* __restrict__ sums, float* __restrict__ counts, int n) {
    int wave = threadIdx.x >> 5, lane = threadIdx.x & 31;
    int node = blockIdx.x * 8 + wave;
    if (node >= n) return;
    int g = batch[node];
    const float4* p = (const float4*)(OUT2 + (size_t)node * HIDC);
    float4 v = p[lane];
    float* sp = sums + (size_t)g * HIDC + lane * 4;
    atomicAdd(sp + 0, fmaxf(v.x, 0.f));
    atomicAdd(sp + 1, fmaxf(v.y, 0.f));
    atomicAdd(sp + 2, fmaxf(v.z, 0.f));
    atomicAdd(sp + 3, fmaxf(v.w, 0.f));
    if (lane == 0) atomicAdd(&counts[g], 1.0f);
}

// ---------------- MLP head: out[g] = relu(mean@W3^T+b3)@W4^T+b4 ----------------
__global__ void k_head(const float* __restrict__ sums, const float* __restrict__ counts,
                       const float* __restrict__ W3, const float* __restrict__ b3,
                       const float* __restrict__ W4, const float* __restrict__ b4,
                       float* __restrict__ out) {
    int g = blockIdx.x;
    __shared__ float mean[HIDC];
    __shared__ float hmid[64];
    float cnt = fmaxf(counts[g], 1.0f);
    for (int c = threadIdx.x; c < HIDC; c += blockDim.x)
        mean[c] = sums[(size_t)g * HIDC + c] / cnt;
    __syncthreads();
    if (threadIdx.x < 64) {
        float acc = b3[threadIdx.x];
        #pragma unroll 8
        for (int c = 0; c < HIDC; ++c) acc += mean[c] * W3[threadIdx.x * HIDC + c];
        hmid[threadIdx.x] = fmaxf(acc, 0.f);
    }
    __syncthreads();
    if (threadIdx.x < 2) {
        float acc = b4[threadIdx.x];
        #pragma unroll 8
        for (int j = 0; j < 64; ++j) acc += hmid[j] * W4[threadIdx.x * 64 + j];
        out[g * 2 + threadIdx.x] = acc;
    }
}

extern "C" void kernel_launch(void* const* d_in, const int* in_sizes, int n_in,
                              void* d_out, int out_size, void* d_ws, size_t ws_size,
                              hipStream_t stream) {
    const int*   x_idx    = (const int*)  d_in[0];
    const int*   eidx     = (const int*)  d_in[1];
    const float* eattr    = (const float*)d_in[2];
    const int*   batch    = (const int*)  d_in[3];
    const float* emb      = (const float*)d_in[4];
    const float* W1       = (const float*)d_in[5];
    const float* asrc1    = (const float*)d_in[6];
    const float* adst1    = (const float*)d_in[7];
    const float* W2       = (const float*)d_in[8];
    const float* asrc2    = (const float*)d_in[9];
    const float* adst2    = (const float*)d_in[10];
    const float* W3       = (const float*)d_in[11];
    const float* b3       = (const float*)d_in[12];
    const float* W4       = (const float*)d_in[13];
    const float* b4       = (const float*)d_in[14];

    const int N  = in_sizes[0];          // 50000 (multiple of 16)
    const int E  = in_sizes[2];          // 600000
    const int TE = E + N;                // edges incl. self loops
    const int G  = 64;
    const int* esrc = eidx;
    const int* edst = eidx + E;

    // ---- carve workspace ----
    size_t off = 0;
    auto carve = [&](size_t bytes) -> void* {
        void* p = (char*)d_ws + off;
        off += (bytes + 255) & ~(size_t)255;
        return p;
    };
    _Float16* Xh     = (_Float16*)carve((size_t)N * HIDC * 2);       // GEMM A (layer input)
    _Float16* Wh     = (_Float16*)carve((size_t)2 * HIDC * HIDC * 2);// f16 W1|W2
    float*    Hbuf   = (float*)   carve((size_t)N * HIDC * 4);       // H = X@W^T
    float*    OUTbuf = (float*)   carve((size_t)N * HIDC * 4);       // aggregated output
    float*    asn    = (float*)   carve((size_t)N * 4);
    float*    adn    = (float*)   carve((size_t)N * 4);
    float*    alphaE = (float*)   carve((size_t)TE * 4);
    unsigned* segmax = (unsigned*)carve((size_t)N * 4);
    float*    segsum = (float*)   carve((size_t)N * 4);
    float*    sums   = (float*)   carve((size_t)G * HIDC * 4);
    float*    counts = (float*)   carve((size_t)G * 4);

    const int T = 256;
    auto blk = [](int n, int t) { return (n + t - 1) / t; };
    const unsigned NEG_INF_ENC = 0x007FFFFFu;   // enc_f(-inf)

    // ---- one-time prep ----
    k_f32_to_f16<<<blk(HIDC * HIDC, T), T, 0, stream>>>(W1, Wh, HIDC * HIDC);
    k_f32_to_f16<<<blk(HIDC * HIDC, T), T, 0, stream>>>(W2, Wh + HIDC * HIDC, HIDC * HIDC);
    k_embed<<<blk(N * HIDC, T), T, 0, stream>>>(x_idx, emb, Xh, N);
    k_fill_f32<<<blk(G * HIDC, T), T, 0, stream>>>(sums, 0.f, G * HIDC);
    k_fill_f32<<<blk(G, T), T, 0, stream>>>(counts, 0.f, G);

    const int gemmBlocks = (N >> 4) * 8 / 8;    // (row tiles * 8 col tiles) / 8 waves per block

    for (int layer = 0; layer < 2; ++layer) {
        const _Float16* Wl = Wh + (size_t)layer * HIDC * HIDC;
        const float* as_v = layer ? asrc2 : asrc1;
        const float* ad_v = layer ? adst2 : adst1;

        // H = X @ W^T  (WMMA)
        k_gemm_wmma<<<gemmBlocks, T, 0, stream>>>(Xh, Wl, Hbuf, N);
        // per-node scores
        k_node_scores<<<blk(N, 8), T, 0, stream>>>(Hbuf, as_v, ad_v, asn, adn, N);
        // softmax state
        k_fill_u32<<<blk(N, T), T, 0, stream>>>(segmax, NEG_INF_ENC, N);
        k_fill_f32<<<blk(N, T), T, 0, stream>>>(segsum, 0.f, N);
        k_fill_f32<<<blk(N * HIDC, T), T, 0, stream>>>(OUTbuf, 0.f, N * HIDC);
        // segment softmax + aggregation
        k_edge_logits<<<blk(TE, T), T, 0, stream>>>(esrc, edst, eattr, asn, adn,
                                                    alphaE, segmax, E, N);
        k_edge_expsum<<<blk(TE, T), T, 0, stream>>>(edst, alphaE, segmax, segsum, E, N);
        k_edge_aggregate<<<blk(TE * 32, T), T, 0, stream>>>(esrc, edst, alphaE, segsum,
                                                            Hbuf, OUTbuf, E, N);
        if (layer == 0) {
            // relu -> f16 input for layer 2
            k_relu_tohalf<<<blk(N * HIDC, T), T, 0, stream>>>(OUTbuf, Xh, N * HIDC);
        } else {
            // relu fused into mean-pool accumulation
            k_pool<<<blk(N, 8), T, 0, stream>>>(OUTbuf, batch, sums, counts, N);
        }
    }

    k_head<<<G, 128, 0, stream>>>(sums, counts, W3, b3, W4, b4, (float*)d_out);
}
